// TransformerUnit_MultiScale_Local_14869176778763
// MI455X (gfx1250) — matrix operand order repaired and checked by reading, not compiled, not run on hardware
//
#include <hip/hip_runtime.h>
#include <hip/hip_bf16.h>
#include <math.h>
#include <stdint.h>

// ---------------- problem constants ----------------
#define BB 8
#define KK 512
#define CC 256
#define HH 8
#define DD 32
#define NN 64
#define FFD 1024
#define MROWS (BB * KK)          // 4096
#define HT 9

typedef __attribute__((ext_vector_type(16))) __bf16 v16bf;
typedef __attribute__((ext_vector_type(8)))  __bf16 v8bf;
typedef __attribute__((ext_vector_type(8)))  float  v8f;
typedef unsigned int v4u __attribute__((ext_vector_type(4)));
typedef int          v4i __attribute__((ext_vector_type(4)));
typedef int          v8i __attribute__((ext_vector_type(8)));

#ifndef __has_builtin
#define __has_builtin(x) 0
#endif
#if __has_builtin(__builtin_amdgcn_tensor_load_to_lds) && \
    __has_builtin(__builtin_amdgcn_s_wait_tensorcnt)
#define USE_TDM 1
#else
#define USE_TDM 0
#endif

__device__ __forceinline__ v8f wmma_bf16(v16bf a, v16bf b, v8f c) {
  return __builtin_amdgcn_wmma_f32_16x16x32_bf16(false, a, false, b, (short)0, c,
                                                 false, false);
}

__device__ __forceinline__ v16bf combine8(v8bf lo, v8bf hi) {
  v16bf a;
#pragma unroll
  for (int j = 0; j < 8; ++j) { a[j] = lo[j]; a[j + 8] = hi[j]; }
  return a;
}

// A fragment: 16x32 bf16 tile, row-major in LDS, leading dim ld (elements).
// lane L: row = L%16 ; holds K = {kh..kh+15}, kh = 16*(L>=16)
__device__ __forceinline__ v16bf frag_a(const __bf16* p, int ld) {
  int lane = threadIdx.x & 31;
  const __bf16* r = p + (lane & 15) * ld + ((lane >> 4) << 4);
  return combine8(*(const v8bf*)r, *(const v8bf*)(r + 8));
}

// B fragment: 32x16 bf16 tile (k-major rows), leading dim ld (strided reads).
__device__ __forceinline__ v16bf frag_b(const __bf16* p, int ld) {
  int lane = threadIdx.x & 31;
  const __bf16* c = p + (lane & 15) + ((lane >> 4) << 4) * ld;
  v16bf b;
#pragma unroll
  for (int j = 0; j < 16; ++j) b[j] = c[j * ld];
  return b;
}

// B fragment from fragment-major swizzled LDS: lane's 16 values contiguous.
__device__ __forceinline__ v16bf frag_b_swz(const __bf16* Bs, int nt) {
  int lane = threadIdx.x & 31;
  const __bf16* r = Bs + nt * 512 + lane * 16;
  return combine8(*(const v8bf*)r, *(const v8bf*)(r + 8));
}

__device__ __forceinline__ float gelu_f(float x) {
  float x3 = x * x * x;
  return 0.5f * x * (1.f + tanhf(0.7978845608028654f * (x + 0.044715f * x3)));
}

// ---------------- elementwise convert ----------------
__global__ __launch_bounds__(256) void f32_to_bf16_kernel(const float* __restrict__ in,
                                                          __bf16* __restrict__ out, int n) {
  int i = blockIdx.x * 256 + threadIdx.x;
  if (i < n) out[i] = (__bf16)in[i];
}

// ---------------- fused Wq*Wk^T per head ----------------
__global__ __launch_bounds__(256) void wfuse_kernel(const float* __restrict__ Wq,
                                                    const float* __restrict__ Wk,
                                                    const float* __restrict__ bq,
                                                    __bf16* __restrict__ Mh,
                                                    float* __restrict__ bqk) {
  int idx = blockIdx.x * 256 + threadIdx.x;        // 8*256*256
  int c2 = idx & 255;
  int t = idx >> 8;
  int c1 = t & 255;
  int h = t >> 8;
  int col = h * DD;
  float s = 0.f;
#pragma unroll 8
  for (int d = 0; d < DD; ++d) s += Wq[c1 * CC + col + d] * Wk[c2 * CC + col + d];
  Mh[idx] = (__bf16)s;
  if (c1 == 0) {
    float sb = 0.f;
#pragma unroll 8
    for (int d = 0; d < DD; ++d) sb += bq[col + d] * Wk[c2 * CC + col + d];
    bqk[h * CC + c2] = sb;
  }
}

// ---------------- rel-pos bilinear grid sample ----------------
__global__ __launch_bounds__(256) void bias_sample_kernel(const float* __restrict__ spat,
                                                          const float* __restrict__ table,
                                                          float* __restrict__ bias) {
  __shared__ float tb[HH * HT * HT];
  for (int i = threadIdx.x; i < HH * HT * HT; i += 256) tb[i] = table[i];
  __syncthreads();
  int idx = blockIdx.x * 256 + threadIdx.x;        // (b,k,n): B*K*N
  int n = idx & (NN - 1);
  int bk = idx >> 6;
  float gx = (spat[(size_t)idx * 2 + 0] + 1.f) * 4.f;
  float gy = (spat[(size_t)idx * 2 + 1] + 1.f) * 4.f;
  float fx0 = floorf(gx), fy0 = floorf(gy);
  int x0 = (int)fx0, y0 = (int)fy0, x1 = x0 + 1, y1 = y0 + 1;
  float wx1 = gx - fx0, wx0 = 1.f - wx1;
  float wy1 = gy - fy0, wy0 = 1.f - wy1;
  bool vx0 = (x0 >= 0) & (x0 <= 8), vx1 = (x1 >= 0) & (x1 <= 8);
  bool vy0 = (y0 >= 0) & (y0 <= 8), vy1 = (y1 >= 0) & (y1 <= 8);
  int cx0 = min(max(x0, 0), 8), cx1 = min(max(x1, 0), 8);
  int cy0 = min(max(y0, 0), 8), cy1 = min(max(y1, 0), 8);
  float w00 = (vy0 && vx0) ? wy0 * wx0 : 0.f;
  float w01 = (vy0 && vx1) ? wy0 * wx1 : 0.f;
  float w10 = (vy1 && vx0) ? wy1 * wx0 : 0.f;
  float w11 = (vy1 && vx1) ? wy1 * wx1 : 0.f;
  int i00 = cy0 * HT + cx0, i01 = cy0 * HT + cx1;
  int i10 = cy1 * HT + cx0, i11 = cy1 * HT + cx1;
#pragma unroll
  for (int h = 0; h < HH; ++h) {
    const float* th = tb + h * HT * HT;
    float v = w00 * th[i00] + w01 * th[i01] + w10 * th[i10] + w11 * th[i11];
    bias[((size_t)bk * HH + h) * NN + n] = v;
  }
}

// ---------------- generic bf16 WMMA GEMM ----------------
// C[M,N] = A[M,K]*B[K,N] + bias ; tile 64x32 per 256-thread block (8 waves)
template <bool GELU>
__global__ __launch_bounds__(256) void gemm_bf16_kernel(
    const __bf16* __restrict__ A, int lda,
    const __bf16* __restrict__ Bm, int ldb,
    const float* __restrict__ bias,
    float* __restrict__ Cf, __bf16* __restrict__ Cb, int ldc, int Kd) {
  __shared__ __attribute__((aligned(16))) __bf16 As[64 * 40];
  __shared__ __attribute__((aligned(16))) __bf16 Bs[1024];   // fragment-major swizzle
  const int tid = threadIdx.x;
  const int m0 = blockIdx.x * 64;
  const int n0 = blockIdx.y * 32;
  const int w = tid >> 5, lane = tid & 31;
  const int mt = w >> 1, nt = w & 1;
  const int arow = tid >> 2, ac = (tid & 3) << 3;
  const int krow = tid >> 3, n4 = (tid & 7) << 2;
  v8f acc = {};
  for (int kt = 0; kt < Kd; kt += 32) {
    const __bf16* ag = A + (size_t)(m0 + arow) * lda + kt + ac;
#pragma unroll
    for (int i = 0; i < 8; ++i) As[arow * 40 + ac + i] = ag[i];
    const __bf16* bg = Bm + (size_t)(kt + krow) * ldb + n0 + n4;
#pragma unroll
    for (int i = 0; i < 4; ++i) {
      int n = n4 + i;
      Bs[((n >> 4) << 9) + ((((krow >> 4) << 4) | (n & 15)) << 4) + (krow & 15)] = bg[i];
    }
    __syncthreads();
    v16bf af = frag_a(&As[(mt * 16) * 40], 40);
    v16bf bf = frag_b_swz(Bs, nt);
    acc = wmma_bf16(af, bf, acc);
    __syncthreads();
  }
  const int col = n0 + nt * 16 + (lane & 15);
  const int row0 = m0 + mt * 16 + ((lane >> 4) << 3);
  const float badd = bias ? bias[col] : 0.f;
#pragma unroll
  for (int r = 0; r < 8; ++r) {
    float v = acc[r] + badd;
    if (GELU) v = gelu_f(v);
    size_t off = (size_t)(row0 + r) * ldc + col;
    if (Cf) Cf[off] = v;
    if (Cb) Cb[off] = (__bf16)v;
  }
}

// ---------------- attention core: one block per (b,k) ----------------
#define LDF 264
#define LDQ 264
#define LDA 72
__global__ __launch_bounds__(256) void attn_core_kernel(
    const float* __restrict__ lf, const __bf16* __restrict__ qk,
    const float* __restrict__ bias, __bf16* __restrict__ fw) {
#if USE_TDM
  __shared__ __attribute__((aligned(16))) float fl[CC * NN];   // TDM dest [c][n] f32
#endif
  __shared__ __attribute__((aligned(16))) __bf16 feats[NN * LDF];
  __shared__ __attribute__((aligned(16))) __bf16 qkl[16 * LDQ];  // rows 8..15 zero
  __shared__ float sc[NN * 16];
  __shared__ __attribute__((aligned(16))) __bf16 at[16 * LDA];
  const int bk = blockIdx.x;                       // b*512+k
  const int b = bk >> 9, k = bk & 511;
  const int tid = threadIdx.x;
  const int w = tid >> 5, lane = tid & 31;
  const size_t tile_elem = ((size_t)b * CC) * (KK * NN) + (size_t)k * NN;

#if USE_TDM
  // Issue the 2-D tensor DMA from wave 0: tile_dim0 = 64 f32 (contiguous n),
  // tile_dim1 = 256 rows (c) with stride K*N elements. LDS dest = fl[c][n].
  if (w == 0) {
    uint64_t gaddr = (uint64_t)(uintptr_t)(lf + tile_elem);
    uint32_t lds_off = (uint32_t)(uintptr_t)(void*)&fl[0];
    v4u g0;
    g0[0] = 1u;                                    // count=1, user descriptor
    g0[1] = lds_off;                               // lds_addr
    g0[2] = (uint32_t)gaddr;                       // global_addr[31:0]
    g0[3] = (uint32_t)((gaddr >> 32) & 0x01FFFFFFu) | (2u << 30);  // addr hi | type=2
    v8i g1;
    g1[0] = 0x00020000;                            // data_size=4B, no multicast
    g1[1] = (int)(64u << 16);                      // tensor_dim0 = 64
    g1[2] = (int)(256u << 16);                     // tensor_dim1 = 256
    g1[3] = (int)(64u << 16);                      // tile_dim0 = 64
    g1[4] = 256;                                   // tile_dim1 = 256
    g1[5] = KK * NN;                               // tensor_dim0_stride = 32768
    g1[6] = 0;
    g1[7] = 0;
    v4i z = {0, 0, 0, 0};
#if __clang_major__ >= 23
    v8i z8 = {0, 0, 0, 0, 0, 0, 0, 0};
    __builtin_amdgcn_tensor_load_to_lds(g0, g1, z, z, z8, 0);
#else
    __builtin_amdgcn_tensor_load_to_lds(g0, g1, z, z, 0);
#endif
  }
#endif

  // stage qk rows as bf16; pad rows 8..15 with zeros (no masked loads later)
  {
    const int h = tid >> 5, c0 = (tid & 31) << 3;
    const __bf16* src = qk + ((size_t)bk * HH + h) * CC + c0;
#pragma unroll
    for (int i = 0; i < 8; ++i) qkl[h * LDQ + c0 + i] = src[i];
#pragma unroll
    for (int i = 0; i < 8; ++i) qkl[(h + 8) * LDQ + c0 + i] = (__bf16)0.f;
  }

#if USE_TDM
  if (w == 0) __builtin_amdgcn_s_wait_tensorcnt(0);
  __syncthreads();
  // convert fl[c][n] f32 -> feats[n][c] bf16
  {
    const int n = tid & 63, cg = tid >> 6;
    for (int c = cg; c < CC; c += 4)
      feats[n * LDF + c] = (__bf16)fl[c * NN + n];
  }
#else
  // fallback: direct coalesced global loads (contiguous over n per c row)
  {
    const int n = tid & 63, cg = tid >> 6;
    for (int c = cg; c < CC; c += 4)
      feats[n * LDF + c] = (__bf16)lf[tile_elem + (size_t)c * (KK * NN) + n];
  }
#endif
  __syncthreads();

  // scores: M=64 (n), N=16 (h pad), K=256 (c) -> waves 0..3 take n-tiles
  if (w < 4) {
    v8f acc = {};
#pragma unroll
    for (int ks = 0; ks < 8; ++ks) {
      v16bf a = frag_a(&feats[(w * 16) * LDF + ks * 32], LDF);
      // B[k=c][col=h] = qkl[h][c]: contiguous along k (h-major storage)
      const __bf16* r = &qkl[(lane & 15) * LDQ + ks * 32 + ((lane >> 4) << 4)];
      v16bf bf = combine8(*(const v8bf*)r, *(const v8bf*)(r + 8));
      acc = wmma_bf16(a, bf, acc);
    }
    const int hcol = lane & 15;
    const int r0 = w * 16 + ((lane >> 4) << 3);
#pragma unroll
    for (int r = 0; r < 8; ++r) sc[(r0 + r) * 16 + hcol] = acc[r];
  }
  __syncthreads();

  // softmax over n (64) per head with rel-pos bias; attn rows 8..15 zero
  if (tid < HH) {
    const int h = tid;
    const float* bp = bias + ((size_t)bk * HH + h) * NN;
    const float scale = 0.17677669529663687f;      // 1/sqrt(32)
    float mx = -1e30f;
    for (int i = 0; i < NN; ++i)
      mx = fmaxf(mx, sc[i * 16 + h] * scale + bp[i]);
    float s = 0.f;
    for (int i = 0; i < NN; ++i) {
      float p = __expf(sc[i * 16 + h] * scale + bp[i] - mx);
      s += p;
      at[h * LDA + i] = (__bf16)p;
    }
    float inv = 1.f / s;
    for (int i = 0; i < NN; ++i)
      at[h * LDA + i] = (__bf16)((float)at[h * LDA + i] * inv);
  } else if (tid < 16) {
    for (int i = 0; i < NN; ++i) at[tid * LDA + i] = (__bf16)0.f;
  }
  __syncthreads();

  // fw = attn[16,64] * feats[64,256] ; 16 c-tiles, 2 per wave
#pragma unroll
  for (int t = 0; t < 2; ++t) {
    const int nt = w * 2 + t;
    v8f acc = {};
#pragma unroll
    for (int ks = 0; ks < 2; ++ks) {
      v16bf a = frag_a(&at[ks * 32], LDA);
      v16bf bf = frag_b(&feats[(ks * 32) * LDF + nt * 16], LDF);
      acc = wmma_bf16(a, bf, acc);
    }
    if (lane < 16) {                               // rows h=0..7 live in lanes<16
      const int c = nt * 16 + (lane & 15);
#pragma unroll
      for (int r = 0; r < 8; ++r)
        fw[((size_t)bk * HH + r) * CC + c] = (__bf16)acc[r];
    }
  }
}

// ---------------- residual add + LayerNorm ----------------
__global__ __launch_bounds__(256) void add_ln_kernel(const float* __restrict__ a,
                                                     const float* __restrict__ b,
                                                     const float* __restrict__ g,
                                                     const float* __restrict__ be,
                                                     float* __restrict__ outf,
                                                     __bf16* __restrict__ outb) {
  __shared__ float red[256];
  const int row = blockIdx.x, tid = threadIdx.x;
  const size_t off = (size_t)row * CC + tid;
  float v = a[off] + b[off];
  red[tid] = v;
  __syncthreads();
  for (int s = 128; s > 0; s >>= 1) {
    if (tid < s) red[tid] += red[tid + s];
    __syncthreads();
  }
  float mu = red[0] * (1.f / CC);
  __syncthreads();
  float d = v - mu;
  red[tid] = d * d;
  __syncthreads();
  for (int s = 128; s > 0; s >>= 1) {
    if (tid < s) red[tid] += red[tid + s];
    __syncthreads();
  }
  float var = red[0] * (1.f / CC);
  float y = d * rsqrtf(var + 1e-5f) * g[tid] + be[tid];
  outf[off] = y;
  if (outb) outb[off] = (__bf16)y;
}

// ---------------- host launch ----------------
extern "C" void kernel_launch(void* const* d_in, const int* in_sizes, int n_in,
                              void* d_out, int out_size, void* d_ws, size_t ws_size,
                              hipStream_t stream) {
  (void)in_sizes; (void)n_in; (void)out_size; (void)ws_size;
  const float* query      = (const float*)d_in[0];
  const float* local_feat = (const float*)d_in[1];
  const float* local_spat = (const float*)d_in[2];
  const float* rel_table  = (const float*)d_in[3];
  const float* Wq = (const float*)d_in[4];
  const float* bq = (const float*)d_in[5];
  const float* Wk = (const float*)d_in[6];
  // d_in[7] = bk: constant over n for each (b,k,h) -> cancels in softmax
  const float* Wv = (const float*)d_in[8];
  const float* bv = (const float*)d_in[9];
  const float* Wo = (const float*)d_in[10];
  const float* bo = (const float*)d_in[11];
  const float* ln1_g = (const float*)d_in[12];
  const float* ln1_b = (const float*)d_in[13];
  const float* W1 = (const float*)d_in[14];
  const float* b1 = (const float*)d_in[15];
  const float* W2 = (const float*)d_in[16];
  const float* b2 = (const float*)d_in[17];
  const float* ln2_g = (const float*)d_in[18];
  const float* ln2_b = (const float*)d_in[19];
  float* out = (float*)d_out;

  char* ws = (char*)d_ws;
  size_t off = 0;
  auto alloc = [&](size_t bytes) {
    size_t o = off;
    off += (bytes + 255) & ~(size_t)255;
    return o;
  };
  __bf16* qbf    = (__bf16*)(ws + alloc((size_t)MROWS * CC * 2));
  __bf16* Mh_bf  = (__bf16*)(ws + alloc((size_t)HH * CC * CC * 2));
  float*  bqk    = (float*) (ws + alloc((size_t)HH * CC * 4));
  __bf16* Wv_bf  = (__bf16*)(ws + alloc((size_t)CC * CC * 2));
  __bf16* Wo_bf  = (__bf16*)(ws + alloc((size_t)CC * CC * 2));
  __bf16* W1_bf  = (__bf16*)(ws + alloc((size_t)CC * FFD * 2));
  __bf16* W2_bf  = (__bf16*)(ws + alloc((size_t)FFD * CC * 2));
  float*  biasw  = (float*) (ws + alloc((size_t)MROWS * HH * NN * 4));
  __bf16* qk_bf  = (__bf16*)(ws + alloc((size_t)MROWS * HH * CC * 2));
  __bf16* fw_bf  = (__bf16*)(ws + alloc((size_t)MROWS * HH * CC * 2));
  __bf16* ctx_bf = (__bf16*)(ws + alloc((size_t)MROWS * CC * 2));
  float*  yf     = (float*) (ws + alloc((size_t)MROWS * CC * 4));
  float*  xf     = (float*) (ws + alloc((size_t)MROWS * CC * 4));
  __bf16* x_bf   = (__bf16*)(ws + alloc((size_t)MROWS * CC * 2));
  __bf16* h_bf   = (__bf16*)(ws + alloc((size_t)MROWS * FFD * 2));
  float*  zf     = (float*) (ws + alloc((size_t)MROWS * CC * 4));

  f32_to_bf16_kernel<<<(MROWS * CC + 255) / 256, 256, 0, stream>>>(query, qbf, MROWS * CC);
  f32_to_bf16_kernel<<<(CC * CC + 255) / 256, 256, 0, stream>>>(Wv, Wv_bf, CC * CC);
  f32_to_bf16_kernel<<<(CC * CC + 255) / 256, 256, 0, stream>>>(Wo, Wo_bf, CC * CC);
  f32_to_bf16_kernel<<<(CC * FFD + 255) / 256, 256, 0, stream>>>(W1, W1_bf, CC * FFD);
  f32_to_bf16_kernel<<<(FFD * CC + 255) / 256, 256, 0, stream>>>(W2, W2_bf, FFD * CC);

  wfuse_kernel<<<(HH * CC * CC) / 256, 256, 0, stream>>>(Wq, Wk, bq, Mh_bf, bqk);
  bias_sample_kernel<<<(MROWS * NN) / 256, 256, 0, stream>>>(local_spat, rel_table, biasw);

  // qk[bk,h,:] = query * (Wq_h Wk_h^T) + bqk   (per-head GEMM, 4096x256x256)
  for (int h = 0; h < HH; ++h)
    gemm_bf16_kernel<false><<<dim3(MROWS / 64, CC / 32), 256, 0, stream>>>(
        qbf, CC, Mh_bf + (size_t)h * CC * CC, CC, bqk + h * CC,
        nullptr, qk_bf + h * CC, HH * CC, CC);

  attn_core_kernel<<<MROWS, 256, 0, stream>>>(local_feat, qk_bf, biasw, fw_bf);

  // ctx[:, h*32:(h+1)*32] = fw_h * Wv_h + bv_h   (4096x32x256)
  for (int h = 0; h < HH; ++h)
    gemm_bf16_kernel<false><<<dim3(MROWS / 64, 1), 256, 0, stream>>>(
        fw_bf + h * CC, HH * CC, Wv_bf + h * DD, CC, bv + h * DD,
        nullptr, ctx_bf + h * DD, CC, CC);

  gemm_bf16_kernel<false><<<dim3(MROWS / 64, CC / 32), 256, 0, stream>>>(
      ctx_bf, CC, Wo_bf, CC, bo, yf, nullptr, CC, CC);

  add_ln_kernel<<<MROWS, 256, 0, stream>>>(query, yf, ln1_g, ln1_b, xf, x_bf);

  gemm_bf16_kernel<true><<<dim3(MROWS / 64, FFD / 32), 256, 0, stream>>>(
      x_bf, CC, W1_bf, FFD, b1, nullptr, h_bf, FFD, CC);

  gemm_bf16_kernel<false><<<dim3(MROWS / 64, CC / 32), 256, 0, stream>>>(
      h_bf, FFD, W2_bf, CC, b2, zf, nullptr, CC, FFD);

  add_ln_kernel<<<MROWS, 256, 0, stream>>>(xf, zf, ln2_g, ln2_b, out, nullptr);
}
